// Transformer_38800734552796
// MI455X (gfx1250) — matrix operand order repaired and checked by reading, not compiled, not run on hardware
//
#include <hip/hip_runtime.h>
#include <hip/hip_bf16.h>

#define B_  8
#define LQ_ 1024
#define LK_ 1024
#define D_  1024
#define H_  8
#define DH_ 128

typedef __bf16 v16bf __attribute__((ext_vector_type(16)));
typedef __bf16 v2bf  __attribute__((ext_vector_type(2)));
typedef float  v8f   __attribute__((ext_vector_type(8)));
typedef int    v4i   __attribute__((vector_size(4 * sizeof(int))));

#if defined(__has_builtin)
#  if __has_builtin(__builtin_amdgcn_global_load_async_to_lds_b128) && \
      __has_builtin(__builtin_amdgcn_s_wait_asynccnt)
#    define HAVE_ASYNC_LDS 1
#  endif
#  if __has_builtin(__builtin_amdgcn_cvt_pk_bf16_f32)
#    define HAVE_PK_BF16 1
#  endif
#endif
#ifndef HAVE_ASYNC_LDS
#  define HAVE_ASYNC_LDS 0
#endif
#ifndef HAVE_PK_BF16
#  define HAVE_PK_BF16 0
#endif

union Frag16 {
  v16bf v;
  uint4 q[2];
};

__device__ __forceinline__ unsigned short f2bf(float f) {
  union { float f; unsigned int u; } x;
  x.f = f;
  unsigned int u = x.u;
  u += 0x7fffu + ((u >> 16) & 1u);   // round-to-nearest-even
  return (unsigned short)(u >> 16);
}

// packed f32x2 -> bf16x2 (one v_cvt_pk_bf16_f32 when available)
__device__ __forceinline__ unsigned int f2bf_pk(float a, float b) {
#if HAVE_PK_BF16
  union { v2bf v; unsigned int u; } x;
  x.v = __builtin_amdgcn_cvt_pk_bf16_f32(a, b);
  return x.u;
#else
  return (unsigned int)f2bf(a) | ((unsigned int)f2bf(b) << 16);
#endif
}

// async memory -> LDS copy of 16 bytes (per lane); falls back to VGPR copy
__device__ __forceinline__ void copy16_g2lds(const unsigned short* gp,
                                             unsigned short* lp) {
#if HAVE_ASYNC_LDS
  __builtin_amdgcn_global_load_async_to_lds_b128(
      (__attribute__((address_space(1))) v4i*)(uintptr_t)gp,
      (__attribute__((address_space(3))) v4i*)(uintptr_t)lp, 0, 0);
#else
  *(uint4*)lp = *(const uint4*)gp;
#endif
}

__device__ __forceinline__ void wait_async_then_barrier() {
#if HAVE_ASYNC_LDS
  __builtin_amdgcn_s_wait_asynccnt(0);
#endif
  __syncthreads();
}

// ---------------------------------------------------------------------------
// Projection GEMM: Y[M,D] = bf16( X[M,D] @ W[D,D] + bias ),  M = B_*LQ_
// Block tile 128x128, 8 waves, each wave: 2(M) x 4(N) WMMA 16x16 tiles, K=32/step
// ---------------------------------------------------------------------------
__global__ __launch_bounds__(256)
void proj_gemm_bf16(const float* __restrict__ X, const float* __restrict__ W,
                    const float* __restrict__ bias, unsigned short* __restrict__ Y) {
  constexpr int AS = 40;    // LDS row stride (ushorts) for A tile 128x32
  constexpr int BS = 136;   // LDS row stride for B tile 32x128
  __shared__ unsigned short As[128 * AS];
  __shared__ unsigned short Bs[32 * BS];

  const int t    = threadIdx.x;
  const int lane = t & 31;
  const int w    = t >> 5;
  const int wm   = w & 3;        // M sub-block: wm*32
  const int wn   = w >> 2;       // N sub-block: wn*64
  const int hi   = lane >> 4;
  const int lo   = lane & 15;
  const int mTile = blockIdx.x * 128;
  const int nTile = blockIdx.y * 128;

  v8f acc[2][4];
#pragma unroll
  for (int mi = 0; mi < 2; ++mi)
#pragma unroll
    for (int ni = 0; ni < 4; ++ni)
#pragma unroll
      for (int e = 0; e < 8; ++e) acc[mi][ni][e] = 0.0f;

  for (int kt = 0; kt < D_ / 32; ++kt) {
    const int kBase = kt * 32;
    // stage A: 128x32 f32 -> bf16 (1024 float4 total, 4 per thread)
#pragma unroll
    for (int i = 0; i < 4; ++i) {
      int idx = t + i * 256;
      int row = idx >> 3;
      int ch  = idx & 7;
      const float4 v = *(const float4*)&X[(size_t)(mTile + row) * D_ + kBase + ch * 4];
      unsigned int* dst = (unsigned int*)&As[row * AS + ch * 4];
      dst[0] = f2bf_pk(v.x, v.y);
      dst[1] = f2bf_pk(v.z, v.w);
    }
    // stage B: 32x128 f32 -> bf16
#pragma unroll
    for (int i = 0; i < 4; ++i) {
      int idx = t + i * 256;
      int row = idx >> 5;
      int ch  = idx & 31;
      const float4 v = *(const float4*)&W[(size_t)(kBase + row) * D_ + nTile + ch * 4];
      unsigned int* dst = (unsigned int*)&Bs[row * BS + ch * 4];
      dst[0] = f2bf_pk(v.x, v.y);
      dst[1] = f2bf_pk(v.z, v.w);
    }
    __syncthreads();

    // A-fragment: lanes 0-15 hold K {kOff..kOff+7, kOff+16..kOff+23}, kOff = hi*8
    Frag16 a[2];
    const int kOff = hi * 8;
#pragma unroll
    for (int mi = 0; mi < 2; ++mi) {
      const unsigned short* ap = &As[(wm * 32 + mi * 16 + lo) * AS];
      a[mi].q[0] = *(const uint4*)&ap[kOff];
      a[mi].q[1] = *(const uint4*)&ap[kOff + 16];
    }
#pragma unroll
    for (int ni = 0; ni < 4; ++ni) {
      // B-fragment: lane = N column lo, K in same order as A-fragment
      Frag16 b;
      const unsigned short* bp = &Bs[wn * 64 + ni * 16 + lo];
#pragma unroll
      for (int j = 0; j < 16; ++j)
        ((unsigned short*)&b)[j] = bp[(j < 8 ? kOff + j : kOff + 8 + j) * BS];
#pragma unroll
      for (int mi = 0; mi < 2; ++mi)
        acc[mi][ni] = __builtin_amdgcn_wmma_f32_16x16x32_bf16(
            false, a[mi].v, false, b.v, (short)0, acc[mi][ni], false, false);
    }
    __syncthreads();
  }

  // epilogue: bias + bf16 store (C layout: elem e -> row e+8*hi, col lo)
#pragma unroll
  for (int ni = 0; ni < 4; ++ni) {
    const int col = nTile + wn * 64 + ni * 16 + lo;
    const float bv = bias[col];
#pragma unroll
    for (int mi = 0; mi < 2; ++mi) {
      const int rbase = mTile + wm * 32 + mi * 16 + hi * 8;
#pragma unroll
      for (int e = 0; e < 8; ++e)
        Y[(size_t)(rbase + e) * D_ + col] = f2bf(acc[mi][ni][e] + bv);
    }
  }
}

// ---------------------------------------------------------------------------
// V = bf16(k)  (V is the raw key input in this model)
// ---------------------------------------------------------------------------
__global__ __launch_bounds__(256)
void cast_bf16(const float* __restrict__ X, unsigned short* __restrict__ Y) {
  const size_t i = ((size_t)blockIdx.x * 256 + threadIdx.x) * 4;
  const float4 v = *(const float4*)&X[i];
  uint2 r;
  r.x = f2bf_pk(v.x, v.y);
  r.y = f2bf_pk(v.z, v.w);
  *(uint2*)&Y[i] = r;
}

// ---------------------------------------------------------------------------
// Flash attention: one block per (b, h, 128 q-rows); 8 waves x 16 q-rows.
// Online softmax over 32-key slabs; scores and P never touch HBM.
// Q/V tiles staged via async global->LDS (no VGPR round trip) when available.
// ---------------------------------------------------------------------------
__global__ __launch_bounds__(256)
void flash_attn(const unsigned short* __restrict__ Qbf,
                const unsigned short* __restrict__ Kbf,
                const unsigned short* __restrict__ Vbf,
                float* __restrict__ att) {
  constexpr int QS = 136, KTS = 40, VS = 136, PS = 40;
  __shared__ unsigned short Qs[128 * QS];   // Q tile  [qrow][dh]
  __shared__ unsigned short Kts[128 * KTS]; // K^T tile [dh][key]
  __shared__ unsigned short Vs[32 * VS];    // V tile  [key][dh]
  __shared__ unsigned short Ps[8 * 16 * PS];// per-wave P staging [qrow][key]

  const int t = threadIdx.x, lane = t & 31, w = t >> 5;
  const int hi = lane >> 4, lo = lane & 15;
  const int qTile = blockIdx.x * 128;
  const int h = blockIdx.y;
  const int b = blockIdx.z;
  const size_t qRowBase = (size_t)b * LQ_;
  const size_t kRowBase = (size_t)b * LK_;
  const int hc = h * DH_;
  const int kOff = hi * 8;

  // stage Q block once (2048 x 16B, 8 per thread) -- async to LDS
#pragma unroll
  for (int i = 0; i < 8; ++i) {
    int idx = t + i * 256;
    int row = idx >> 4;
    int ch  = idx & 15;
    copy16_g2lds(&Qbf[(qRowBase + qTile + row) * D_ + hc + ch * 8],
                 &Qs[row * QS + ch * 8]);
  }
  wait_async_then_barrier();

  // preload loop-invariant Q A-fragments (wave's 16 rows, 4 dh-chunks of 32)
  Frag16 aQ[4];
#pragma unroll
  for (int dc = 0; dc < 4; ++dc) {
    const unsigned short* qp = &Qs[(w * 16 + lo) * QS + dc * 32];
    aQ[dc].q[0] = *(const uint4*)&qp[kOff];
    aQ[dc].q[1] = *(const uint4*)&qp[kOff + 16];
  }

  v8f o[8];
  float m[8], l[8];
#pragma unroll
  for (int n = 0; n < 8; ++n)
#pragma unroll
    for (int e = 0; e < 8; ++e) o[n][e] = 0.0f;
#pragma unroll
  for (int e = 0; e < 8; ++e) { m[e] = -__builtin_inff(); l[e] = 0.0f; }

  const float scale = 0.088388347648318447f; // 1/sqrt(DH)

  for (int kt = 0; kt < LK_ / 32; ++kt) {
    __syncthreads(); // all waves done with previous Kts/Vs
    // stage K^T tile: read [key][dh] bf16 rows, scatter to [dh][key]
#pragma unroll
    for (int i = 0; i < 2; ++i) {
      int idx = t + i * 256;
      int key = idx >> 4;
      int ch  = idx & 15;
      union { uint4 q; unsigned short s[8]; } u;
      u.q = *(const uint4*)&Kbf[(kRowBase + kt * 32 + key) * D_ + hc + ch * 8];
#pragma unroll
      for (int j = 0; j < 8; ++j)
        Kts[(ch * 8 + j) * KTS + key] = u.s[j];
      if (kt + 1 < LK_ / 32)   // pull next K tile toward L2 / WGP$
        __builtin_prefetch(&Kbf[(kRowBase + (kt + 1) * 32 + key) * D_ + hc + ch * 8], 0, 1);
    }
    // stage V tile: straight copy -> async global->LDS
#pragma unroll
    for (int i = 0; i < 2; ++i) {
      int idx = t + i * 256;
      int key = idx >> 4;
      int ch  = idx & 15;
      copy16_g2lds(&Vbf[(kRowBase + kt * 32 + key) * D_ + hc + ch * 8],
                   &Vs[key * VS + ch * 8]);
    }
    wait_async_then_barrier();

    // scores: two 16-key tiles, each accumulated over 4 WMMAs (dh = 4x32)
    v8f s[2];
#pragma unroll
    for (int kn = 0; kn < 2; ++kn) {
#pragma unroll
      for (int e = 0; e < 8; ++e) s[kn][e] = 0.0f;
#pragma unroll
      for (int dc = 0; dc < 4; ++dc) {
        Frag16 bK;
        const unsigned short* kp = &Kts[(dc * 32) * KTS + kn * 16 + lo];
#pragma unroll
        for (int j = 0; j < 16; ++j)
          ((unsigned short*)&bK)[j] = kp[(j < 8 ? kOff + j : kOff + 8 + j) * KTS];
        s[kn] = __builtin_amdgcn_wmma_f32_16x16x32_bf16(
            false, aQ[dc].v, false, bK.v, (short)0, s[kn], false, false);
      }
#pragma unroll
      for (int e = 0; e < 8; ++e) s[kn][e] *= scale;
    }

    // online softmax over this 32-key slab (rows live in 16-lane half-groups)
    float alpha[8];
#pragma unroll
    for (int e = 0; e < 8; ++e) {
      float v = fmaxf(s[0][e], s[1][e]);
#pragma unroll
      for (int mask = 1; mask <= 8; mask <<= 1)
        v = fmaxf(v, __shfl_xor(v, mask, 32));
      const float mn = fmaxf(m[e], v);
      alpha[e] = __expf(m[e] - mn);
      m[e] = mn;
      s[0][e] = __expf(s[0][e] - mn);
      s[1][e] = __expf(s[1][e] - mn);
      float rs = s[0][e] + s[1][e];
#pragma unroll
      for (int mask = 1; mask <= 8; mask <<= 1)
        rs += __shfl_xor(rs, mask, 32);
      l[e] = l[e] * alpha[e] + rs;
    }
#pragma unroll
    for (int n = 0; n < 8; ++n)
#pragma unroll
      for (int e = 0; e < 8; ++e) o[n][e] *= alpha[e];

    // reshape P: C-layout -> A-fragment layout via per-wave LDS (no barrier:
    // same wave, DS ops ordered by DScnt)
    unsigned short* pw = &Ps[w * 16 * PS];
#pragma unroll
    for (int kn = 0; kn < 2; ++kn)
#pragma unroll
      for (int e = 0; e < 8; ++e)
        pw[(e + 8 * hi) * PS + kn * 16 + lo] = f2bf(s[kn][e]);
    Frag16 aP;
    {
      const unsigned short* pp = &pw[lo * PS];
      aP.q[0] = *(const uint4*)&pp[kOff];
      aP.q[1] = *(const uint4*)&pp[kOff + 16];
    }

    // O += P @ V  (8 dh-column tiles of 16)
#pragma unroll
    for (int n = 0; n < 8; ++n) {
      Frag16 bV;
      const unsigned short* vp = &Vs[n * 16 + lo];
#pragma unroll
      for (int j = 0; j < 16; ++j)
        ((unsigned short*)&bV)[j] = vp[(j < 8 ? kOff + j : kOff + 8 + j) * VS];
      o[n] = __builtin_amdgcn_wmma_f32_16x16x32_bf16(
          false, aP.v, false, bV.v, (short)0, o[n], false, false);
    }
  }

  // normalize + store fp32 attention output
#pragma unroll
  for (int n = 0; n < 8; ++n)
#pragma unroll
    for (int e = 0; e < 8; ++e) {
      const int row = qTile + w * 16 + e + 8 * hi;
      const int col = hc + n * 16 + lo;
      att[(qRowBase + row) * D_ + col] = o[n][e] / l[e];
    }
}

// ---------------------------------------------------------------------------
// Residual blend + LayerNorm: one block per (b, q-row); 256 threads x 4 cols
// ---------------------------------------------------------------------------
__global__ __launch_bounds__(256)
void blend_ln(const float* __restrict__ att, const float* __restrict__ q,
              const float* __restrict__ gamma, const float* __restrict__ beta,
              float* __restrict__ out) {
  __shared__ float red[16];
  const int t = threadIdx.x;
  const size_t base = (size_t)blockIdx.x * D_;
  const int c = t * 4;

  const float4 a = *(const float4*)&att[base + c];
  const float4 x = *(const float4*)&q[base + c];
  float y0 = a.x * 0.1f + x.x * 0.9f;
  float y1 = a.y * 0.1f + x.y * 0.9f;
  float y2 = a.z * 0.1f + x.z * 0.9f;
  float y3 = a.w * 0.1f + x.w * 0.9f;

  float s  = y0 + y1 + y2 + y3;
  float sq = y0 * y0 + y1 * y1 + y2 * y2 + y3 * y3;
#pragma unroll
  for (int mask = 1; mask < 32; mask <<= 1) {
    s  += __shfl_xor(s,  mask, 32);
    sq += __shfl_xor(sq, mask, 32);
  }
  const int w = t >> 5, lane = t & 31;
  if (lane == 0) { red[w] = s; red[8 + w] = sq; }
  __syncthreads();
  if (t < 32) {
    float ss = (t < 8) ? red[t] : 0.0f;
    float qq = (t < 8) ? red[8 + t] : 0.0f;
#pragma unroll
    for (int mask = 1; mask < 8; mask <<= 1) {
      ss += __shfl_xor(ss, mask, 32);
      qq += __shfl_xor(qq, mask, 32);
    }
    if (t == 0) { red[0] = ss; red[1] = qq; }
  }
  __syncthreads();
  const float mu  = red[0] * (1.0f / D_);
  const float var = red[1] * (1.0f / D_) - mu * mu;
  const float inv = rsqrtf(var + 1e-5f);

  const float4 g  = *(const float4*)&gamma[c];
  const float4 bb = *(const float4*)&beta[c];
  float4 r;
  r.x = (y0 - mu) * inv * g.x + bb.x;
  r.y = (y1 - mu) * inv * g.y + bb.y;
  r.z = (y2 - mu) * inv * g.z + bb.z;
  r.w = (y3 - mu) * inv * g.w + bb.w;
  *(float4*)&out[base + c] = r;
}

// ---------------------------------------------------------------------------
extern "C" void kernel_launch(void* const* d_in, const int* in_sizes, int n_in,
                              void* d_out, int out_size, void* d_ws, size_t ws_size,
                              hipStream_t stream) {
  const float* q  = (const float*)d_in[0];
  const float* k  = (const float*)d_in[1];
  const float* Wq = (const float*)d_in[2];
  const float* bq = (const float*)d_in[3];
  const float* Wk = (const float*)d_in[4];
  const float* bk = (const float*)d_in[5];
  const float* g  = (const float*)d_in[6];
  const float* be = (const float*)d_in[7];
  float* out = (float*)d_out;

  char* ws = (char*)d_ws;
  unsigned short* Qbf = (unsigned short*)(ws);                              // 16 MiB
  unsigned short* Kbf = (unsigned short*)(ws + (size_t)16 * 1024 * 1024);   // 16 MiB
  unsigned short* Vbf = (unsigned short*)(ws + (size_t)32 * 1024 * 1024);   // 16 MiB
  float*          att = (float*)        (ws + (size_t)48 * 1024 * 1024);    // 32 MiB

  const dim3 gproj(64, 8);                  // (M/128, D/128)
  proj_gemm_bf16<<<gproj, 256, 0, stream>>>(q, Wq, bq, Qbf);
  proj_gemm_bf16<<<gproj, 256, 0, stream>>>(k, Wk, bk, Kbf);
  cast_bf16<<<(B_ * LK_ * D_) / 1024, 256, 0, stream>>>(k, Vbf);

  const dim3 gattn(LQ_ / 128, H_, B_);
  flash_attn<<<gattn, 256, 0, stream>>>(Qbf, Kbf, Vbf, att);

  blend_ln<<<B_ * LQ_, 256, 0, stream>>>(att, q, g, be, out);
}
